// Attention_71167608095073
// MI455X (gfx1250) — compile-verified, matrix operand order checked by reading, not run on hardware
//
#include <hip/hip_runtime.h>
#include <hip/hip_bf16.h>
#include <math.h>

// ---------------------------------------------------------------------------
// Problem constants (B=1)
// ---------------------------------------------------------------------------
#define N_TOK   2048
#define DIM     1024
#define NHEAD   16
#define DH      64
#define ROT     32
#define MEM     16
#define NBUCKET 32
#define REL_SCALE 8.0f            // sqrt(DH)
#define NPAD    2080              // 16 pad rows + MEM + N  (divisible by 32)
#define MASKVAL (-3.402823466e38f)

typedef __attribute__((ext_vector_type(16))) _Float16 v16h;
typedef __attribute__((ext_vector_type(8)))  float    v8f;
typedef __attribute__((ext_vector_type(2)))  _Float16 h2v;

union AFrag { v16h v; _Float16 h[16]; unsigned int u[8]; };
union CFrag { v8f  v; float     f[8]; };

// A/B 16-bit fragment K-index base for VGPR r (ISA 7.12.2, 16-bit A 16x32):
// element pair (2r, 2r+1) holds K = kbase(r,grp) + {0,1}
__device__ __forceinline__ int kbase(int r, int grp) {
    return (r < 4) ? (grp * 8 + 2 * r) : (16 + grp * 8 + 2 * (r - 4));
}

__device__ __forceinline__ v8f wmma_f16(const AFrag& a, const AFrag& b, v8f c) {
    return __builtin_amdgcn_wmma_f32_16x16x32_f16(false, a.v, false, b.v,
                                                  (short)0, c, false, false);
}

// load a 16x32 fragment from a contiguous f16 row (pairs -> dword/b128 loads)
__device__ __forceinline__ AFrag load_frag(const _Float16* row, int off, int grp) {
    AFrag f;
#pragma unroll
    for (int r = 0; r < 8; ++r)
        f.u[r] = *(const unsigned int*)(row + off + kbase(r, grp));
    return f;
}

// pack two f32 -> dword of two f16 (v_cvt_pk_f16_f32)
__device__ __forceinline__ unsigned int packh2(float a, float b) {
    union { _Float16 h[2]; unsigned int u; } t;
    t.h[0] = (_Float16)a; t.h[1] = (_Float16)b;
    return t.u;
}

// relative position bucket (matches _rel_bucket: max_exact=16, log base 8)
__device__ __forceinline__ int rel_bucket(int n) {
    if (n <= 0)  return 0;
    if (n < 16)  return n;
    int v = 16 + (int)(__logf((float)n * (1.0f / 16.0f)) * (16.0f / 2.0794415417f));
    return v < 31 ? v : 31;
}

// stage a 16x64 f32 tile (rows stride `srcStride`) into row-major f16 LDS
// 128 threads: one uint4 (b128) store per thread
__device__ __forceinline__ void stage_a16x64(
    const float* __restrict__ src, int srcStride, _Float16* lds, int tid)
{
    const int t8 = tid * 8, row = t8 >> 6, col = t8 & 63;
    const float4* p = (const float4*)&src[row * srcStride + col];
    const float4 v0 = p[0], v1 = p[1];
    uint4 pk;
    pk.x = packh2(v0.x, v0.y); pk.y = packh2(v0.z, v0.w);
    pk.z = packh2(v1.x, v1.y); pk.w = packh2(v1.z, v1.w);
    *(uint4*)&lds[row * 64 + col] = pk;
}

// stage a 64(k) x 64(n) f32 tile into f16 LDS transposed [n][k]
// two k-rows per task so (k,k+1) pairs pack into ds_store_b32
__device__ __forceinline__ void stage_bT64x64(
    const float* __restrict__ src, int srcStride, _Float16* lds, int tid)
{
#pragma unroll
    for (int j = 0; j < 4; ++j) {
        const int task = tid + 128 * j;         // 0..511
        const int k    = (task >> 4) * 2;       // even k row
        const int c4   = (task & 15) * 4;       // n
        const float4 va = *(const float4*)&src[(k + 0) * srcStride + c4];
        const float4 vb = *(const float4*)&src[(k + 1) * srcStride + c4];
        *(unsigned int*)&lds[(c4 + 0) * 64 + k] = packh2(va.x, vb.x);
        *(unsigned int*)&lds[(c4 + 1) * 64 + k] = packh2(va.y, vb.y);
        *(unsigned int*)&lds[(c4 + 2) * 64 + k] = packh2(va.z, vb.z);
        *(unsigned int*)&lds[(c4 + 3) * 64 + k] = packh2(va.w, vb.w);
    }
}

// ---------------------------------------------------------------------------
// Kernel 1: QKV projections  x(2048x1024) @ W(1024x1024)  -> raw f32
// grid (16, 128, 3)  block 128 (4 waves).  LDS-staged tiles:
//   A: 16x64 f16 (x rows), B: 64x64 f16 stored transposed [n][k]
// ---------------------------------------------------------------------------
__global__ __launch_bounds__(128) void qkv_gemm(
    const float* __restrict__ x,  const float* __restrict__ Wq,
    const float* __restrict__ Wk, const float* __restrict__ Wv,
    float* __restrict__ qraw, float* __restrict__ kraw, float* __restrict__ vraw)
{
    const int tid  = threadIdx.x;
    const int lane = tid & 31, wv = tid >> 5;
    const int grp = lane >> 4, nl = lane & 15;
    const int i0 = blockIdx.y * 16;
    const int c0blk = blockIdx.x * 64;
    const int c0 = c0blk + wv * 16;
    const float* W   = (blockIdx.z == 0) ? Wq : (blockIdx.z == 1 ? Wk : Wv);
    float*       Out = (blockIdx.z == 0) ? qraw : (blockIdx.z == 1 ? kraw : vraw);

    __shared__ _Float16 ldsA[16 * 64];      // [m][k]
    __shared__ _Float16 ldsB[64 * 64];      // [n][k]  (transposed)

    CFrag acc;
#pragma unroll
    for (int r = 0; r < 8; ++r) acc.f[r] = 0.0f;

    for (int kk = 0; kk < DIM; kk += 64) {
        stage_a16x64(&x[i0 * DIM + kk], DIM, ldsA, tid);
        stage_bT64x64(&W[kk * DIM + c0blk], DIM, ldsB, tid);
        __syncthreads();
#pragma unroll
        for (int c = 0; c < 2; ++c) {
            const AFrag a = load_frag(&ldsA[nl * 64], c * 32, grp);
            const AFrag b = load_frag(&ldsB[(wv * 16 + nl) * 64], c * 32, grp);
            acc.v = wmma_f16(a, b, acc.v);
        }
        __syncthreads();
    }
#pragma unroll
    for (int r = 0; r < 8; ++r)
        Out[(i0 + r + 8 * grp) * DIM + c0 + nl] = acc.f[r];
}

// ---------------------------------------------------------------------------
// Kernel 2: rotary(q,k,v) + l2norm(q,k) + mem-slot concat + f16 pack
// ---------------------------------------------------------------------------
__device__ __forceinline__ float blk_sum64(float v, float* red, int d) {
    red[d] = v; __syncthreads();
#pragma unroll
    for (int s = 32; s > 0; s >>= 1) {
        if (d < s) red[d] += red[d + s];
        __syncthreads();
    }
    float r = red[0]; __syncthreads();
    return r;
}

__global__ __launch_bounds__(64) void rope_pack(
    const float* __restrict__ qraw, const float* __restrict__ kraw,
    const float* __restrict__ vraw, const float* __restrict__ rope,
    const float* __restrict__ mem_k, const float* __restrict__ mem_v,
    _Float16* __restrict__ Qh, _Float16* __restrict__ Kh, _Float16* __restrict__ Vh)
{
    const int jp = blockIdx.x, h = blockIdx.y, d = threadIdx.x;
    __shared__ float sq[64], sk[64], sv[64], red[64];

    _Float16* Krow = Kh + ((size_t)h * NPAD + jp) * DH;
    _Float16* Vrow = Vh + ((size_t)h * NPAD + jp) * DH;

    if (jp < 16) {                       // pad rows -> zero (masked later)
        Krow[d] = (_Float16)0.0f; Vrow[d] = (_Float16)0.0f; return;
    }
    if (jp < 32) {                       // memory slots (k normalized, v raw)
        const int m = jp - 16;
        float kv = mem_k[(h * MEM + m) * DH + d];
        float vv = mem_v[(h * MEM + m) * DH + d];
        float nk = blk_sum64(kv * kv, red, d);
        nk = fmaxf(sqrtf(nk), 1e-12f);
        Krow[d] = (_Float16)(kv / nk);
        Vrow[d] = (_Float16)vv;
        return;
    }
    const int t = jp - 32;               // real token
    float qv = qraw[t * DIM + h * DH + d];
    float kv = kraw[t * DIM + h * DH + d];
    float vv = vraw[t * DIM + h * DH + d];
    sq[d] = qv; sk[d] = kv; sv[d] = vv;
    __syncthreads();
    if (d < ROT) {                       // rotary on first 32 dims (q,k,v)
        const float f = rope[t * ROT + d];
        const float c = __cosf(f), s = __sinf(f);
        const float rq = (d < 16) ? -sq[d + 16] : sq[d - 16];
        const float rk = (d < 16) ? -sk[d + 16] : sk[d - 16];
        const float rv = (d < 16) ? -sv[d + 16] : sv[d - 16];
        qv = qv * c + rq * s; kv = kv * c + rk * s; vv = vv * c + rv * s;
    }
    __syncthreads();
    float nq = blk_sum64(qv * qv, red, d);
    float nk = blk_sum64(kv * kv, red, d);
    nq = fmaxf(sqrtf(nq), 1e-12f);
    nk = fmaxf(sqrtf(nk), 1e-12f);
    Qh[((size_t)h * N_TOK + t) * DH + d] = (_Float16)(qv / nq);
    Krow[d] = (_Float16)(kv / nk);
    Vrow[d] = (_Float16)vv;
}

// ---------------------------------------------------------------------------
// Kernel 3: attention pass 1 — row max / sum for the pre_proj-mixed heads
// grid 128 i-tiles, block 512 (16 waves = all heads)
// ---------------------------------------------------------------------------
__global__ __launch_bounds__(512) void attn_pass1(
    const _Float16* __restrict__ Qh, const _Float16* __restrict__ Kh,
    const float* __restrict__ pre, const float* __restrict__ qk_scale,
    const float* __restrict__ rel_table,
    float* __restrict__ Mbuf, float* __restrict__ Lbuf)
{
    const int lane = threadIdx.x & 31, wv = threadIdx.x >> 5;
    const int grp = lane >> 4, nl = lane & 15;
    const int i0 = blockIdx.x * 16;
    __shared__ float raw[NHEAD][256];                 // 16 KB

    const _Float16* Q = Qh + (size_t)wv * N_TOK * DH;
    const AFrag a0 = load_frag(&Q[(i0 + nl) * DH], 0,  grp);
    const AFrag a1 = load_frag(&Q[(i0 + nl) * DH], 32, grp);

    // fold qk_scale into the pre_proj mix (this wave is mixed head kh = wv)
    float coef[NHEAD];
#pragma unroll
    for (int h = 0; h < NHEAD; ++h)
        coef[h] = pre[h * NHEAD + wv] * (1.0f / fmaxf(__expf(qk_scale[h]), 0.01f));

    float m8[8], l8[8];
#pragma unroll
    for (int r = 0; r < 8; ++r) { m8[r] = -INFINITY; l8[r] = 0.0f; }

    const _Float16* K = Kh + (size_t)wv * NPAD * DH;

    for (int jt = 16; jt < NPAD; jt += 16) {
        if (jt > i0 + 47) break;                      // causal early exit
        if (jt + 16 < NPAD) __builtin_prefetch(&K[(jt + 16 + nl) * DH], 0, 0);

        const AFrag b0 = load_frag(&K[(jt + nl) * DH], 0,  grp);
        const AFrag b1 = load_frag(&K[(jt + nl) * DH], 32, grp);
        CFrag dacc;
#pragma unroll
        for (int r = 0; r < 8; ++r) dacc.f[r] = 0.0f;
        dacc.v = wmma_f16(a0, b0, dacc.v);
        dacc.v = wmma_f16(a1, b1, dacc.v);
#pragma unroll
        for (int r = 0; r < 8; ++r)
            raw[wv][(r + 8 * grp) * 16 + nl] = dacc.f[r];
        __syncthreads();

        // mixed head kh = wv : mix + bias + mask, online softmax stats
#pragma unroll
        for (int r = 0; r < 8; ++r) {
            const int m  = r + 8 * grp;
            const int qi = i0 + m;
            const int jr = jt - 16 + nl;              // reference key index
            float dsum = 0.0f;
#pragma unroll
            for (int h = 0; h < NHEAD; ++h)
                dsum += coef[h] * raw[h][m * 16 + nl];
            dsum += rel_table[rel_bucket(qi - jr) * NHEAD + wv] * REL_SCALE;
            if (jr >= MEM && (jr - MEM) > qi) dsum = MASKVAL;

            float tmax = dsum;
#pragma unroll
            for (int off = 1; off < 16; off <<= 1)
                tmax = fmaxf(tmax, __shfl_xor(tmax, off, 32));
            const float newm = fmaxf(m8[r], tmax);
            float p = __expf(dsum - newm);
#pragma unroll
            for (int off = 1; off < 16; off <<= 1)
                p += __shfl_xor(p, off, 32);
            l8[r] = l8[r] * __expf(m8[r] - newm) + p;
            m8[r] = newm;
        }
        __syncthreads();
    }
    if (nl == 0) {
#pragma unroll
        for (int r = 0; r < 8; ++r) {
            const int row = i0 + r + 8 * grp;
            Mbuf[wv * N_TOK + row] = m8[r];
            Lbuf[wv * N_TOK + row] = l8[r];
        }
    }
}

// ---------------------------------------------------------------------------
// Kernel 4: attention pass 2 — normalized P, post_proj mix, P~ @ V
// grid 128 i-tiles, block 512 (16 waves)
// ---------------------------------------------------------------------------
__global__ __launch_bounds__(512) void attn_pass2(
    const _Float16* __restrict__ Qh, const _Float16* __restrict__ Kh,
    const _Float16* __restrict__ Vh,
    const float* __restrict__ pre, const float* __restrict__ post,
    const float* __restrict__ qk_scale, const float* __restrict__ rel_table,
    const float* __restrict__ Mbuf, const float* __restrict__ Lbuf,
    float* __restrict__ Oh)
{
    const int lane = threadIdx.x & 31, wv = threadIdx.x >> 5;
    const int grp = lane >> 4, nl = lane & 15;
    const int i0 = blockIdx.x * 16;
    __shared__ float    raw[NHEAD][512];              // 32 KB  [h][m*32+nn]
    __shared__ _Float16 Pn [NHEAD][512];              // 16 KB  [kh][m*32+nn]

    const _Float16* Q = Qh + (size_t)wv * N_TOK * DH;
    const AFrag a0 = load_frag(&Q[(i0 + nl) * DH], 0,  grp);
    const AFrag a1 = load_frag(&Q[(i0 + nl) * DH], 32, grp);

    float coef[NHEAD], pcoef[NHEAD];
#pragma unroll
    for (int h = 0; h < NHEAD; ++h) {
        coef[h]  = pre[h * NHEAD + wv] * (1.0f / fmaxf(__expf(qk_scale[h]), 0.01f));
        pcoef[h] = post[h * NHEAD + wv];              // out head ko = wv
    }
    float m8[8], il8[8];
#pragma unroll
    for (int r = 0; r < 8; ++r) {                     // stats of mixed head wv
        const int row = i0 + r + 8 * grp;
        m8[r]  = Mbuf[wv * N_TOK + row];
        il8[r] = 1.0f / Lbuf[wv * N_TOK + row];
    }
    CFrag acc[4];
#pragma unroll
    for (int c = 0; c < 4; ++c)
#pragma unroll
        for (int r = 0; r < 8; ++r) acc[c].f[r] = 0.0f;

    const _Float16* K = Kh + (size_t)wv * NPAD * DH;
    const _Float16* V = Vh + (size_t)wv * NPAD * DH;

    for (int jt2 = 0; jt2 < NPAD; jt2 += 32) {
        if (jt2 > i0 + 47) break;
        if (jt2 + 32 < NPAD) __builtin_prefetch(&V[(jt2 + 32 + nl) * DH], 0, 0);

        // (1) raw dots, input head wv, two 16-key subtiles
#pragma unroll
        for (int sub = 0; sub < 2; ++sub) {
            const int js = jt2 + sub * 16;
            const AFrag b0 = load_frag(&K[(js + nl) * DH], 0,  grp);
            const AFrag b1 = load_frag(&K[(js + nl) * DH], 32, grp);
            CFrag dacc;
#pragma unroll
            for (int r = 0; r < 8; ++r) dacc.f[r] = 0.0f;
            dacc.v = wmma_f16(a0, b0, dacc.v);
            dacc.v = wmma_f16(a1, b1, dacc.v);
#pragma unroll
            for (int r = 0; r < 8; ++r)
                raw[wv][(r + 8 * grp) * 32 + sub * 16 + nl] = dacc.f[r];
        }
        __syncthreads();

        // (2) normalized probabilities for mixed head wv
#pragma unroll
        for (int r = 0; r < 8; ++r) {
            const int m  = r + 8 * grp;
            const int qi = i0 + m;
#pragma unroll
            for (int sub = 0; sub < 2; ++sub) {
                const int nn = sub * 16 + nl;
                const int jr = jt2 + nn - 16;
                float p;
                if (jr < 0 || (jr >= MEM && (jr - MEM) > qi)) {
                    p = 0.0f;
                } else {
                    float dsum = 0.0f;
#pragma unroll
                    for (int h = 0; h < NHEAD; ++h)
                        dsum += coef[h] * raw[h][m * 32 + nn];
                    dsum += rel_table[rel_bucket(qi - jr) * NHEAD + wv] * REL_SCALE;
                    p = __expf(dsum - m8[r]) * il8[r];
                }
                Pn[wv][m * 32 + nn] = (_Float16)p;
            }
        }
        __syncthreads();

        // (3) out head wv : P~ = sum_kh post[kh][wv] * P_kh ; acc += P~ @ V
        AFrag pf;
#pragma unroll
        for (int r = 0; r < 8; ++r) {
            const int kb = kbase(r, grp);             // nn pair base 0..30
            float s0 = 0.0f, s1 = 0.0f;
#pragma unroll
            for (int kh = 0; kh < NHEAD; ++kh) {
                const h2v p = *(const h2v*)&Pn[kh][nl * 32 + kb];
                s0 += pcoef[kh] * (float)p.x;
                s1 += pcoef[kh] * (float)p.y;
            }
            pf.u[r] = packh2(s0, s1);
        }
#pragma unroll
        for (int nc = 0; nc < 4; ++nc) {
            AFrag bv;
#pragma unroll
            for (int r = 0; r < 8; ++r) {
                const int kb = kbase(r, grp);         // key row within step
                bv.h[2 * r]     = V[(jt2 + kb) * DH + nc * 16 + nl];
                bv.h[2 * r + 1] = V[(jt2 + kb + 1) * DH + nc * 16 + nl];
            }
            acc[nc].v = wmma_f16(pf, bv, acc[nc].v);
        }
        __syncthreads();
    }
#pragma unroll
    for (int nc = 0; nc < 4; ++nc)
#pragma unroll
        for (int r = 0; r < 8; ++r)
            Oh[((size_t)wv * N_TOK + i0 + r + 8 * grp) * DH + nc * 16 + nl]
                = acc[nc].f[r];
}

// ---------------------------------------------------------------------------
// Kernel 5: out = Ocat(2048x1024) @ Wo + bo   (LDS-staged like qkv_gemm)
// Each 64-wide K chunk is exactly one head slice of Oh (contiguous rows).
// ---------------------------------------------------------------------------
__global__ __launch_bounds__(128) void out_proj(
    const float* __restrict__ Oh, const float* __restrict__ Wo,
    const float* __restrict__ bo, float* __restrict__ out)
{
    const int tid  = threadIdx.x;
    const int lane = tid & 31, wv = tid >> 5;
    const int grp = lane >> 4, nl = lane & 15;
    const int i0 = blockIdx.y * 16;
    const int c0blk = blockIdx.x * 64;
    const int c0 = c0blk + wv * 16;

    __shared__ _Float16 ldsA[16 * 64];      // [m][k]
    __shared__ _Float16 ldsB[64 * 64];      // [n][k]  (transposed)

    CFrag acc;
#pragma unroll
    for (int r = 0; r < 8; ++r) acc.f[r] = 0.0f;

    for (int kk = 0; kk < DIM; kk += 64) {
        const int hh = kk >> 6;             // head for this feature chunk
        stage_a16x64(&Oh[((size_t)hh * N_TOK + i0) * DH], DH, ldsA, tid);
        stage_bT64x64(&Wo[kk * DIM + c0blk], DIM, ldsB, tid);
        __syncthreads();
#pragma unroll
        for (int c = 0; c < 2; ++c) {
            const AFrag a = load_frag(&ldsA[nl * 64], c * 32, grp);
            const AFrag b = load_frag(&ldsB[(wv * 16 + nl) * 64], c * 32, grp);
            acc.v = wmma_f16(a, b, acc.v);
        }
        __syncthreads();
    }
    const float bias = bo[c0 + nl];
#pragma unroll
    for (int r = 0; r < 8; ++r)
        out[(i0 + r + 8 * grp) * DIM + c0 + nl] = acc.f[r] + bias;
}

// ---------------------------------------------------------------------------
// Host launcher
// ---------------------------------------------------------------------------
extern "C" void kernel_launch(void* const* d_in, const int* in_sizes, int n_in,
                              void* d_out, int out_size, void* d_ws, size_t ws_size,
                              hipStream_t stream)
{
    const float* x       = (const float*)d_in[0];
    const float* rope    = (const float*)d_in[1];
    const float* Wq      = (const float*)d_in[2];
    const float* Wk      = (const float*)d_in[3];
    const float* Wv      = (const float*)d_in[4];
    const float* Wo      = (const float*)d_in[5];
    const float* bo      = (const float*)d_in[6];
    const float* mem_k   = (const float*)d_in[7];
    const float* mem_v   = (const float*)d_in[8];
    const float* pre     = (const float*)d_in[9];
    const float* post    = (const float*)d_in[10];
    const float* qkscale = (const float*)d_in[11];
    const float* rel_tab = (const float*)d_in[12];
    float* out = (float*)d_out;

    char* ws = (char*)d_ws;
    const size_t SZ_RAW = (size_t)N_TOK * DIM * 4;               // 8 MB
    float*    qraw = (float*)(ws);                               // aliased w/ Oh
    float*    kraw = (float*)(ws + SZ_RAW);
    float*    vraw = (float*)(ws + 2 * SZ_RAW);
    _Float16* Qh   = (_Float16*)(ws + 3 * SZ_RAW);
    _Float16* Kh   = (_Float16*)(ws + 3 * SZ_RAW + (size_t)NHEAD * N_TOK * DH * 2);
    _Float16* Vh   = (_Float16*)((char*)Kh + (size_t)NHEAD * NPAD * DH * 2);
    float*    Mb   = (float*)((char*)Vh + (size_t)NHEAD * NPAD * DH * 2);
    float*    Lb   = (float*)((char*)Mb + (size_t)NHEAD * N_TOK * 4);
    float*    Oh   = (float*)(ws);   // reuse qraw region (consumed by rope_pack)

    qkv_gemm<<<dim3(DIM / 64, N_TOK / 16, 3), 128, 0, stream>>>(
        x, Wq, Wk, Wv, qraw, kraw, vraw);

    rope_pack<<<dim3(NPAD, NHEAD), 64, 0, stream>>>(
        qraw, kraw, vraw, rope, mem_k, mem_v, Qh, Kh, Vh);

    attn_pass1<<<N_TOK / 16, 512, 0, stream>>>(
        Qh, Kh, pre, qkscale, rel_tab, Mb, Lb);

    attn_pass2<<<N_TOK / 16, 512, 0, stream>>>(
        Qh, Kh, Vh, pre, post, qkscale, rel_tab, Mb, Lb, Oh);

    out_proj<<<dim3(DIM / 64, N_TOK / 16), 128, 0, stream>>>(
        Oh, Wo, bo, out);
}
